// WaveGPT_2654289789229
// MI455X (gfx1250) — compile-verified, matrix operand order checked by reading, not run on hardware
//
#include <hip/hip_runtime.h>
#include <math.h>

// ---------------- model dims ----------------
#define V_SZ  32000
#define D_SZ  768
#define H_SZ  12
#define NW_SZ 16
#define NH_SZ 4
#define L_SZ  4
#define B_SZ  4
#define T_SZ  1024
#define DF_SZ 3072
#define HD_SZ 64
#define M_ROWS (B_SZ * T_SZ)   // 4096 token rows

typedef __attribute__((ext_vector_type(16))) __bf16          v16bf;
typedef __attribute__((ext_vector_type(16))) unsigned short  v16u;
typedef __attribute__((ext_vector_type(8)))  float           v8f;

#if defined(__has_builtin)
#  if __has_builtin(__builtin_amdgcn_global_load_async_to_lds_b128)
#    define HAVE_ASYNC_LDS 1
#  endif
#endif
#ifndef HAVE_ASYNC_LDS
#  define HAVE_ASYNC_LDS 0
#endif

static __device__ __forceinline__ unsigned short f2bf(float f) {
  unsigned int u = __builtin_bit_cast(unsigned int, f);
  u += 0x7fffu + ((u >> 16) & 1u);           // round-to-nearest-even
  return (unsigned short)(u >> 16);
}
static __device__ __forceinline__ float bf2f(unsigned short h) {
  unsigned int u = ((unsigned int)h) << 16;
  return __builtin_bit_cast(float, u);
}

// Fragment assembly from two 16B chunks (frag element order is two
// contiguous 8-element K runs: e0..7 -> K=kh*8..+7, e8..15 -> K=kh*8+16..+23)
static __device__ __forceinline__ v16u ld_frag_us(const unsigned short* p0,
                                                  const unsigned short* p1) {
  union { uint4 q[2]; v16u v; } u;
  u.q[0] = *(const uint4*)p0;
  u.q[1] = *(const uint4*)p1;
  return u.v;
}
static __device__ __forceinline__ v16u ld_frag_u(const unsigned int* p) {
  union { uint4 q[2]; v16u v; } u;
  u.q[0] = *(const uint4*)p;
  u.q[1] = *(const uint4*)(p + 8);
  return u.v;
}
static __device__ __forceinline__ v16u ld_frag_half(const unsigned short* p0) {
  union { uint4 q[2]; v16u v; } u;
  u.q[0] = *(const uint4*)p0;
  uint4 z; z.x = 0u; z.y = 0u; z.z = 0u; z.w = 0u;
  u.q[1] = z;
  return u.v;
}

// =====================================================================
// Weight pre-pack: W(KxN fp32) -> Wp[K/32][N][32] bf16 (K-ascending per
// (tile,col) chunk == (k,k+1) dword pairs for the WMMA B fragment).
// =====================================================================
__global__ __launch_bounds__(256)
void prepack_w_kernel(const float* __restrict__ W, unsigned short* __restrict__ Wp,
                      int K, int N) {
  int idx = blockIdx.x * blockDim.x + threadIdx.x;
  int nkt = K >> 5;
  if (idx >= nkt * N) return;
  int kt = idx / N;
  int c  = idx - kt * N;
  union { unsigned short us[32]; uint4 q[4]; } buf;
  const float* src = W + (size_t)kt * 32 * N + c;
  for (int k = 0; k < 32; ++k)
    buf.us[k] = f2bf(src[(size_t)k * N]);
  uint4* dst = (uint4*)(Wp + (((size_t)kt * N + c) << 5));
  dst[0] = buf.q[0]; dst[1] = buf.q[1]; dst[2] = buf.q[2]; dst[3] = buf.q[3];
}

// =====================================================================
// Wave-packet embedding: one block per token row
// =====================================================================
__global__ __launch_bounds__(256)
void embed_kernel(const int* __restrict__ tok,
                  const float* __restrict__ base_f, const float* __restrict__ harm_a,
                  const float* __restrict__ phases, const float* __restrict__ pos_f,
                  const float* __restrict__ w2e_w,  const float* __restrict__ w2e_b,
                  const float* __restrict__ simple, const float* __restrict__ wave_ratio,
                  const float* __restrict__ ln_g,   const float* __restrict__ ln_b,
                  float* __restrict__ xout) {
  const int row = blockIdx.x;              // b*T + t
  const int t   = row % T_SZ;
  const int tid = threadIdx.x;
  const int tk  = tok[row];

  __shared__ float wfeat[2 * NW_SZ * NH_SZ];   // 128 features
  __shared__ float red[256], red2[256];

  if (tid < NW_SZ * NH_SZ) {
    int n = tid / NH_SZ, j = tid % NH_SZ;
    float bf = base_f[tk * NW_SZ + n];
    float ph = phases[tk * NW_SZ + n];
    float pp = (float)t * pos_f[n];
    float wp = bf * (float)(j + 1) * 6.2831853071795864f + ph + pp;
    float ha = harm_a[(tk * NW_SZ + n) * NH_SZ + j];
    wfeat[tid]      = ha * sinf(wp);
    wfeat[64 + tid] = ha * cosf(wp);
  }
  __syncthreads();

  const float ratio = 1.0f / (1.0f + expf(-wave_ratio[0]));
  float vals[3], s = 0.f, s2 = 0.f;
  for (int q = 0; q < 3; ++q) {
    int d = tid + q * 256;
    float acc = 0.f;
    for (int f = 0; f < 128; ++f) acc += wfeat[f] * w2e_w[f * D_SZ + d];
    float v = ratio * (acc + w2e_b[d]) + (1.f - ratio) * simple[(size_t)tk * D_SZ + d];
    vals[q] = v; s += v; s2 += v * v;
  }
  red[tid] = s; red2[tid] = s2;
  __syncthreads();
  for (int st = 128; st > 0; st >>= 1) {
    if (tid < st) { red[tid] += red[tid + st]; red2[tid] += red2[tid + st]; }
    __syncthreads();
  }
  float mu   = red[0] / (float)D_SZ;
  float var  = red2[0] / (float)D_SZ - mu * mu;
  float rstd = rsqrtf(var + 1e-5f);
  for (int q = 0; q < 3; ++q) {
    int d = tid + q * 256;
    xout[(size_t)row * D_SZ + d] = (vals[q] - mu) * rstd * ln_g[d] + ln_b[d];
  }
}

// =====================================================================
// LayerNorm fp32 -> bf16 row kernel
// =====================================================================
__global__ __launch_bounds__(256)
void ln_bf16_kernel(const float* __restrict__ x, const float* __restrict__ g,
                    const float* __restrict__ b, unsigned short* __restrict__ out) {
  const int row = blockIdx.x;
  const int tid = threadIdx.x;
  __shared__ float red[256], red2[256];
  float vals[3], s = 0.f, s2 = 0.f;
  for (int q = 0; q < 3; ++q) {
    float v = x[(size_t)row * D_SZ + tid + q * 256];
    vals[q] = v; s += v; s2 += v * v;
  }
  red[tid] = s; red2[tid] = s2;
  __syncthreads();
  for (int st = 128; st > 0; st >>= 1) {
    if (tid < st) { red[tid] += red[tid + st]; red2[tid] += red2[tid + st]; }
    __syncthreads();
  }
  float mu   = red[0] / (float)D_SZ;
  float var  = red2[0] / (float)D_SZ - mu * mu;
  float rstd = rsqrtf(var + 1e-5f);
  for (int q = 0; q < 3; ++q) {
    int d = tid + q * 256;
    out[(size_t)row * D_SZ + d] = f2bf((vals[q] - mu) * rstd * g[d] + b[d]);
  }
}

// =====================================================================
// GEMM: C(MxN) = A_bf16(MxK) @ Wp_bf16packed(KxN) [+bias] [+resid]
//  - 128x128x32 tiles, 8 waves, 2x4 WMMA frags per wave
//  - A staged row-major (frag reads = 2 x b128, K-contiguous runs)
//  - W pre-packed [K/32][N][32]: staging & frag reads are pure b128 ops
//  - double-buffered LDS; next tile staged via CDNA5 async global->LDS
//    DMA (ASYNCcnt) when available, else sync stage at top of iteration.
//    Either way no staging registers live across the WMMA region.
// =====================================================================
#define BM 128
#define BN 128
#define BK 32
#define APAD 40   // ushorts per A row   (80B: 16B aligned, spread banks)
#define BPAD 20   // uints per B column  (80B: 16B aligned, spread banks)

#if HAVE_ASYNC_LDS
// builtin signature (from clang diagnostic): (v4i AS1*, v4i AS3*, Imm, Imm)
typedef int v4i_gcc __attribute__((vector_size(16)));
typedef __attribute__((address_space(1))) v4i_gcc* g_v4i_p;
typedef __attribute__((address_space(3))) v4i_gcc* l_v4i_p;
static __device__ __forceinline__ void async_copy16(const void* g, void* l) {
  __builtin_amdgcn_global_load_async_to_lds_b128((g_v4i_p)g, (l_v4i_p)l, 0, 0);
}
static __device__ __forceinline__ void async_wait0() {
#if __has_builtin(__builtin_amdgcn_s_wait_asynccnt)
  __builtin_amdgcn_s_wait_asynccnt(0);
#else
  asm volatile("s_wait_asynccnt 0x0" ::: "memory");
#endif
}
#endif

static __device__ __forceinline__ void stage_tiles(
    const unsigned short* __restrict__ A, const unsigned short* __restrict__ Wp,
    unsigned short (*sAb)[APAD], unsigned int (*sBb)[BPAD],
    int N, int K, int blkM, int blkN, int kt, int tid) {
  const int idx  = tid >> 1;       // A row / B col
  const int half = tid & 1;
  const unsigned short* gA = A + (size_t)(blkM + idx) * K + kt * 32 + half * 16;
  unsigned short* lA = &sAb[idx][half * 16];
  const int gc = blkN + idx;
  unsigned int* lB = &sBb[idx][half * 8];
#if HAVE_ASYNC_LDS
  async_copy16(gA, lA);
  async_copy16(gA + 8, lA + 8);
  if (gc < N) {
    const unsigned short* gB = Wp + (((size_t)kt * N + gc) << 5) + half * 16;
    async_copy16(gB, lB);
    async_copy16(gB + 8, lB + 4);
  } else {
    uint4 z; z.x = 0u; z.y = 0u; z.z = 0u; z.w = 0u;
    ((uint4*)lB)[0] = z;
    *(uint4*)(lB + 4) = z;
  }
#else
  uint4 a0 = ((const uint4*)gA)[0];
  uint4 a1 = ((const uint4*)gA)[1];
  uint4 b0, b1;
  if (gc < N) {
    const uint4* gB = (const uint4*)(Wp + (((size_t)kt * N + gc) << 5) + half * 16);
    b0 = gB[0]; b1 = gB[1];
  } else {
    b0.x = 0u; b0.y = 0u; b0.z = 0u; b0.w = 0u; b1 = b0;
  }
  ((uint4*)lA)[0] = a0;
  ((uint4*)lA)[1] = a1;
  ((uint4*)lB)[0] = b0;
  *(uint4*)(lB + 4) = b1;
#endif
}

__global__ __launch_bounds__(256, 1)
void gemm_bf16_kernel(const unsigned short* __restrict__ A,
                      const unsigned short* __restrict__ Wp,
                      const float* __restrict__ bias,
                      const float* __restrict__ resid,
                      float* __restrict__ Cf,
                      unsigned short* __restrict__ Cb,
                      int M, int N, int K) {
  __shared__ alignas(16) unsigned short sAds[2][BM][APAD];  // 20 KB
  __shared__ alignas(16) unsigned int   sBds[2][BN][BPAD];  // 20 KB

  const int tid   = threadIdx.x;
  const int lane  = tid & 31;
  const int wave  = tid >> 5;
  const int wm    = wave & 3;          // M quarter (32 rows)
  const int wn    = wave >> 2;         // N half    (64 cols)
  const int blkM  = blockIdx.y * BM;
  const int blkN  = blockIdx.x * BN;
  const int lrow  = lane & 15;
  const int khalf = lane >> 4;

  v8f acc[2][4];
  const v8f z8 = {0.f, 0.f, 0.f, 0.f, 0.f, 0.f, 0.f, 0.f};
  for (int i = 0; i < 2; ++i)
    for (int j = 0; j < 4; ++j) acc[i][j] = z8;

  const int nk = K / BK;
  stage_tiles(A, Wp, sAds[0], sBds[0], N, K, blkM, blkN, 0, tid);
#if HAVE_ASYNC_LDS
  async_wait0();
#endif
  __syncthreads();

  for (int kt = 0; kt < nk; ++kt) {
    const int cur = kt & 1;
    // stage next tile into the buffer NOT read this iteration (its last
    // readers finished before the previous barrier)
    if (kt + 1 < nk)
      stage_tiles(A, Wp, sAds[cur ^ 1], sBds[cur ^ 1], N, K, blkM, blkN, kt + 1, tid);

    unsigned short (*cA)[APAD] = sAds[cur];
    unsigned int   (*cB)[BPAD] = sBds[cur];
    v16u af[2], bfm[4];
    for (int tm = 0; tm < 2; ++tm) {
      int rr = wm * 32 + tm * 16 + lrow;
      af[tm] = ld_frag_us(&cA[rr][khalf * 8], &cA[rr][khalf * 8 + 16]);
    }
    for (int tn = 0; tn < 4; ++tn) {
      int c = wn * 64 + tn * 16 + lrow;
      bfm[tn] = ld_frag_u(&cB[c][khalf * 4]);
    }
    for (int tm = 0; tm < 2; ++tm)
      for (int tn = 0; tn < 4; ++tn)
        acc[tm][tn] = __builtin_amdgcn_wmma_f32_16x16x32_bf16(
            false, __builtin_bit_cast(v16bf, af[tm]),
            false, __builtin_bit_cast(v16bf, bfm[tn]),
            (short)0, acc[tm][tn], false, false);

#if HAVE_ASYNC_LDS
    async_wait0();
#endif
    __syncthreads();
  }

  // epilogue: C layout -> lane = col, vgpr r = row (+8 for lanes 16..31)
  for (int tm = 0; tm < 2; ++tm)
    for (int tn = 0; tn < 4; ++tn) {
      int gm0 = blkM + wm * 32 + tm * 16 + khalf * 8;
      int gn  = blkN + wn * 64 + tn * 16 + lrow;
      if (gn >= N) continue;
      float bv = bias[gn];
      for (int rr = 0; rr < 8; ++rr) {
        int gm = gm0 + rr;
        if (gm >= M) continue;
        float v = acc[tm][tn][rr] + bv;
        if (resid) v += resid[(size_t)gm * N + gn];
        if (Cb) Cb[(size_t)gm * N + gn] = f2bf(v);
        else    Cf[(size_t)gm * N + gn] = v;
      }
    }
}

// =====================================================================
// In-place activation on bf16: m = gelu(m, exact erf) + 0.1*sin(m)
// =====================================================================
__global__ void act_bf16_kernel(unsigned short* __restrict__ m, long n) {
  long i = (long)blockIdx.x * blockDim.x + threadIdx.x;
  long stride = (long)gridDim.x * blockDim.x;
  for (; i < n; i += stride) {
    float x = bf2f(m[i]);
    float g = 0.5f * x * (1.f + erff(x * 0.70710678118654752f));
    m[i] = f2bf(g + 0.1f * sinf(x));
  }
}

// =====================================================================
// q/k wave features: qn = l2norm(sin(qf*t + qp)) over NW -> bf16 (B*H,T,NW)
// =====================================================================
__global__ __launch_bounds__(256)
void wave_qk_kernel(const float* __restrict__ qf, const float* __restrict__ qp,
                    const float* __restrict__ kf, const float* __restrict__ kp,
                    unsigned short* __restrict__ qn, unsigned short* __restrict__ kn) {
  int idx = blockIdx.x * blockDim.x + threadIdx.x;     // (b*T+t)*H + h
  if (idx >= M_ROWS * H_SZ) return;
  int h   = idx % H_SZ;
  int row = idx / H_SZ;            // b*T + t
  int t   = row % T_SZ;
  int b   = row / T_SZ;
  float tf = (float)t;
  size_t inoff  = (size_t)row * (H_SZ * NW_SZ) + h * NW_SZ;
  size_t outoff = ((size_t)(b * H_SZ + h) * T_SZ + t) * NW_SZ;

  float sv[NW_SZ]; float nrm = 0.f;
  for (int n = 0; n < NW_SZ; ++n) {
    float s = sinf(qf[inoff + n] * tf + qp[inoff + n]);
    sv[n] = s; nrm += s * s;
  }
  float inv = 1.f / fmaxf(sqrtf(nrm), 1e-12f);
  for (int n = 0; n < NW_SZ; ++n) qn[outoff + n] = f2bf(sv[n] * inv);

  nrm = 0.f;
  for (int n = 0; n < NW_SZ; ++n) {
    float s = sinf(kf[inoff + n] * tf + kp[inoff + n]);
    sv[n] = s; nrm += s * s;
  }
  inv = 1.f / fmaxf(sqrtf(nrm), 1e-12f);
  for (int n = 0; n < NW_SZ; ++n) kn[outoff + n] = f2bf(sv[n] * inv);
}

// =====================================================================
// Fused linear wave attention: per (b,h,16-query tile) stream key-tile
// pairs; S via WMMA (K=16 padded), mask+scale in C layout, S tile to
// LDS (row-major, K-contiguous -> A-frag reads are 2 x b128),
// O += S(16x32) @ V(32x64) at full K=32.
// =====================================================================
__global__ __launch_bounds__(32)
void wave_attn_kernel(const unsigned short* __restrict__ qn,   // (B*H,T,NW)
                      const unsigned short* __restrict__ kn,   // (B*H,T,NW)
                      const unsigned short* __restrict__ vb,   // (B*T,D) bf16
                      const float* __restrict__ att_scale,     // (H)
                      unsigned short* __restrict__ aout) {     // (B*T,D) bf16
  const int qt   = blockIdx.x;          // query tile 0..63
  const int bh   = blockIdx.y;          // 0..B*H-1
  const int b    = bh / H_SZ, h = bh % H_SZ;
  const int lane = threadIdx.x;
  const int lrow = lane & 15;
  const int khalf = lane >> 4;

  __shared__ alignas(16) unsigned short sS[16][40];   // 16 q x 32 k (+pad)
  __shared__ alignas(16) unsigned short sV[32][72];   // 32 k x 64 hd (+pad)

  const v16u aQ =
      ld_frag_half(qn + ((size_t)bh * T_SZ + qt * 16 + lrow) * NW_SZ + khalf * 8);

  const v8f z8 = {0.f, 0.f, 0.f, 0.f, 0.f, 0.f, 0.f, 0.f};
  v8f oacc[4];
  for (int i = 0; i < 4; ++i) oacc[i] = z8;
  const float sc = att_scale[h];

  for (int sbase = 0; sbase <= qt * 16; sbase += 32) {
    v8f S[2];
    for (int hf = 0; hf < 2; ++hf) {
      v16u bK = ld_frag_half(
          kn + ((size_t)bh * T_SZ + sbase + hf * 16 + lrow) * NW_SZ + khalf * 8);
      S[hf] = __builtin_amdgcn_wmma_f32_16x16x32_bf16(
          false, __builtin_bit_cast(v16bf, aQ),
          false, __builtin_bit_cast(v16bf, bK),
          (short)0, z8, false, false);
    }
    // mask (causal) + att_scale * rsqrt(t+1); write S (bf16) to LDS
    for (int rr = 0; rr < 8; ++rr) {
      int tq = qt * 16 + khalf * 8 + rr;
      float f = sc * rsqrtf((float)(tq + 1));
      int ts0 = sbase + lrow;
      int ts1 = sbase + 16 + lrow;
      sS[khalf * 8 + rr][lrow]      = f2bf(ts0 <= tq ? S[0][rr] * f : 0.f);
      sS[khalf * 8 + rr][16 + lrow] = f2bf(ts1 <= tq ? S[1][rr] * f : 0.f);
    }
    // stage V pair-tile 32x64
    for (int idx = lane; idx < 32 * 64; idx += 32) {
      int rr = idx >> 6, c = idx & 63;
      sV[rr][c] = vb[((size_t)(b * T_SZ + sbase + rr)) * D_SZ + h * HD_SZ + c];
    }
    __syncthreads();
    // O += S(16x32) @ V(32x64)
    v16u aS = ld_frag_us(&sS[lrow][khalf * 8], &sS[lrow][khalf * 8 + 16]);
    for (int nt = 0; nt < 4; ++nt) {
      v16u bV;
      for (int e = 0; e < 16; ++e)
        bV[e] = sV[khalf * 8 + (e & 7) + ((e & 8) << 1)][nt * 16 + lrow];
      oacc[nt] = __builtin_amdgcn_wmma_f32_16x16x32_bf16(
          false, __builtin_bit_cast(v16bf, aS),
          false, __builtin_bit_cast(v16bf, bV),
          (short)0, oacc[nt], false, false);
    }
    __syncthreads();
  }
  for (int nt = 0; nt < 4; ++nt)
    for (int rr = 0; rr < 8; ++rr) {
      int tq  = qt * 16 + khalf * 8 + rr;
      int col = h * HD_SZ + nt * 16 + lrow;
      aout[((size_t)(b * T_SZ + tq)) * D_SZ + col] = f2bf(oacc[nt][rr]);
    }
}

// =====================================================================
// Host orchestration
// =====================================================================
extern "C" void kernel_launch(void* const* d_in, const int* in_sizes, int n_in,
                              void* d_out, int out_size, void* d_ws, size_t ws_size,
                              hipStream_t stream) {
  (void)in_sizes; (void)n_in; (void)out_size; (void)ws_size;

  const int*   token_ids      = (const int*)  d_in[0];
  const float* emb_base_freqs = (const float*)d_in[1];
  const float* emb_harm_amps  = (const float*)d_in[2];
  const float* emb_phases     = (const float*)d_in[3];
  const float* emb_pos_freq   = (const float*)d_in[4];
  const float* emb_w2e_w      = (const float*)d_in[5];
  const float* emb_w2e_b      = (const float*)d_in[6];
  const float* emb_simple     = (const float*)d_in[7];
  const float* emb_wave_ratio = (const float*)d_in[8];
  const float* emb_ln_g       = (const float*)d_in[9];
  const float* emb_ln_b       = (const float*)d_in[10];
  const float* ln1_g = (const float*)d_in[11];
  const float* ln1_b = (const float*)d_in[12];
  const float* qf_w  = (const float*)d_in[13];
  const float* qf_b  = (const float*)d_in[14];
  const float* kf_w  = (const float*)d_in[15];
  const float* kf_b  = (const float*)d_in[16];
  const float* qp_w  = (const float*)d_in[17];
  const float* qp_b  = (const float*)d_in[18];
  const float* kp_w  = (const float*)d_in[19];
  const float* kp_b  = (const float*)d_in[20];
  const float* v_w   = (const float*)d_in[21];
  const float* v_b   = (const float*)d_in[22];
  const float* o_w   = (const float*)d_in[23];
  const float* o_b   = (const float*)d_in[24];
  const float* att_scale = (const float*)d_in[25];
  const float* ln2_g = (const float*)d_in[26];
  const float* ln2_b = (const float*)d_in[27];
  const float* fc1_w = (const float*)d_in[28];
  const float* fc1_b = (const float*)d_in[29];
  const float* fc2_w = (const float*)d_in[30];
  const float* fc2_b = (const float*)d_in[31];
  const float* head_ln_g = (const float*)d_in[32];
  const float* head_ln_b = (const float*)d_in[33];
  const float* head_w    = (const float*)d_in[34];
  const float* head_b    = (const float*)d_in[35];

  // workspace layout
  char* wsb = (char*)d_ws;
  size_t off = 0;
  auto alloc = [&](size_t bytes) -> void* {
    off = (off + 255) & ~(size_t)255;
    void* p = wsb + off;
    off += bytes;
    return p;
  };
  const int QN = H_SZ * NW_SZ;   // 192
  float*          xbuf = (float*)alloc((size_t)M_ROWS * D_SZ * 4);
  unsigned short* hbuf = (unsigned short*)alloc((size_t)M_ROWS * D_SZ * 2);
  float*          qf_o = (float*)alloc((size_t)M_ROWS * QN * 4);
  float*          kf_o = (float*)alloc((size_t)M_ROWS * QN * 4);
  float*          qp_o = (float*)alloc((size_t)M_ROWS * QN * 4);
  float*          kp_o = (float*)alloc((size_t)M_ROWS * QN * 4);
  unsigned short* qnb  = (unsigned short*)alloc((size_t)B_SZ * H_SZ * T_SZ * NW_SZ * 2);
  unsigned short* knb  = (unsigned short*)alloc((size_t)B_SZ * H_SZ * T_SZ * NW_SZ * 2);
  unsigned short* vbf  = (unsigned short*)alloc((size_t)M_ROWS * D_SZ * 2);
  unsigned short* aobf = (unsigned short*)alloc((size_t)M_ROWS * D_SZ * 2);
  unsigned short* mbf  = (unsigned short*)alloc((size_t)M_ROWS * DF_SZ * 2);
  // packed bf16 weights (tile-major transposed layout)
  unsigned short* pqf = (unsigned short*)alloc((size_t)L_SZ * D_SZ * QN * 2);
  unsigned short* pkf = (unsigned short*)alloc((size_t)L_SZ * D_SZ * QN * 2);
  unsigned short* pqp = (unsigned short*)alloc((size_t)L_SZ * D_SZ * QN * 2);
  unsigned short* pkp = (unsigned short*)alloc((size_t)L_SZ * D_SZ * QN * 2);
  unsigned short* pv  = (unsigned short*)alloc((size_t)L_SZ * D_SZ * D_SZ * 2);
  unsigned short* po  = (unsigned short*)alloc((size_t)L_SZ * D_SZ * D_SZ * 2);
  unsigned short* pf1 = (unsigned short*)alloc((size_t)L_SZ * D_SZ * DF_SZ * 2);
  unsigned short* pf2 = (unsigned short*)alloc((size_t)L_SZ * DF_SZ * D_SZ * 2);
  unsigned short* phw = (unsigned short*)alloc((size_t)D_SZ * V_SZ * 2);

  auto prepack = [&](const float* W, unsigned short* Wp, int K, int N) {
    int total = (K >> 5) * N;
    prepack_w_kernel<<<(total + 255) / 256, 256, 0, stream>>>(W, Wp, K, N);
  };
  prepack(qf_w, pqf, L_SZ * D_SZ, QN);
  prepack(kf_w, pkf, L_SZ * D_SZ, QN);
  prepack(qp_w, pqp, L_SZ * D_SZ, QN);
  prepack(kp_w, pkp, L_SZ * D_SZ, QN);
  prepack(v_w,  pv,  L_SZ * D_SZ, D_SZ);
  prepack(o_w,  po,  L_SZ * D_SZ, D_SZ);
  prepack(fc1_w, pf1, L_SZ * D_SZ, DF_SZ);
  prepack(fc2_w, pf2, L_SZ * DF_SZ, D_SZ);
  prepack(head_w, phw, D_SZ, V_SZ);

  auto gemm = [&](const unsigned short* A, const unsigned short* Wp, const float* bias,
                  const float* resid, float* Cf, unsigned short* Cb,
                  int M, int N, int K) {
    dim3 grid((N + BN - 1) / BN, (M + BM - 1) / BM);
    gemm_bf16_kernel<<<grid, 256, 0, stream>>>(A, Wp, bias, resid, Cf, Cb, M, N, K);
  };

  // ---- embedding ----
  embed_kernel<<<M_ROWS, 256, 0, stream>>>(
      token_ids, emb_base_freqs, emb_harm_amps, emb_phases, emb_pos_freq,
      emb_w2e_w, emb_w2e_b, emb_simple, emb_wave_ratio, emb_ln_g, emb_ln_b, xbuf);

  // ---- layers ----
  for (int l = 0; l < L_SZ; ++l) {
    ln_bf16_kernel<<<M_ROWS, 256, 0, stream>>>(xbuf, ln1_g + l * D_SZ, ln1_b + l * D_SZ, hbuf);

    gemm(hbuf, pqf + (size_t)l * D_SZ * QN, qf_b + (size_t)l * QN, nullptr, qf_o, nullptr, M_ROWS, QN, D_SZ);
    gemm(hbuf, pkf + (size_t)l * D_SZ * QN, kf_b + (size_t)l * QN, nullptr, kf_o, nullptr, M_ROWS, QN, D_SZ);
    gemm(hbuf, pqp + (size_t)l * D_SZ * QN, qp_b + (size_t)l * QN, nullptr, qp_o, nullptr, M_ROWS, QN, D_SZ);
    gemm(hbuf, pkp + (size_t)l * D_SZ * QN, kp_b + (size_t)l * QN, nullptr, kp_o, nullptr, M_ROWS, QN, D_SZ);
    gemm(hbuf, pv + (size_t)l * D_SZ * D_SZ, v_b + (size_t)l * D_SZ, nullptr, nullptr, vbf, M_ROWS, D_SZ, D_SZ);

    wave_qk_kernel<<<(M_ROWS * H_SZ + 255) / 256, 256, 0, stream>>>(qf_o, qp_o, kf_o, kp_o, qnb, knb);

    dim3 agrid(T_SZ / 16, B_SZ * H_SZ);
    wave_attn_kernel<<<agrid, 32, 0, stream>>>(qnb, knb, vbf, att_scale + (size_t)l * H_SZ, aobf);

    // x = x + o @ o_w + o_b  (residual fused in gemm epilogue)
    gemm(aobf, po + (size_t)l * D_SZ * D_SZ, o_b + (size_t)l * D_SZ, xbuf, xbuf, nullptr, M_ROWS, D_SZ, D_SZ);

    ln_bf16_kernel<<<M_ROWS, 256, 0, stream>>>(xbuf, ln2_g + l * D_SZ, ln2_b + l * D_SZ, hbuf);
    gemm(hbuf, pf1 + (size_t)l * D_SZ * DF_SZ, fc1_b + (size_t)l * DF_SZ, nullptr, nullptr, mbf, M_ROWS, DF_SZ, D_SZ);
    act_bf16_kernel<<<4096, 256, 0, stream>>>(mbf, (long)M_ROWS * DF_SZ);
    gemm(mbf, pf2 + (size_t)l * DF_SZ * D_SZ, fc2_b + (size_t)l * D_SZ, xbuf, xbuf, nullptr, M_ROWS, D_SZ, DF_SZ);
  }

  // ---- head ----
  ln_bf16_kernel<<<M_ROWS, 256, 0, stream>>>(xbuf, head_ln_g, head_ln_b, hbuf);
  gemm(hbuf, phw, head_b, nullptr, (float*)d_out, nullptr, M_ROWS, V_SZ, D_SZ);
}
